// HMNet_8572754722841
// MI455X (gfx1250) — matrix operand order repaired and checked by reading, not compiled, not run on hardware
//
#include <hip/hip_runtime.h>
#include <cstddef>
#include <cstdint>

// ---------------- CDNA5 (gfx1250) wave32 WMMA Swin implementation ----------------
// bf16 activations + weights, f32 accumulation via v_wmma_f32_16x16x32_bf16.
// A-tiles staged by the Tensor Data Mover (tensor_load_to_lds + s_wait_tensorcnt),
// attention q/k/v staged by global_load_async_to_lds_b128 (+ s_wait_asynccnt).

typedef __attribute__((ext_vector_type(16))) __bf16 v16bf;
typedef __attribute__((ext_vector_type(8)))  float  v8f;
typedef __attribute__((ext_vector_type(4))) unsigned v4u;
typedef __attribute__((ext_vector_type(8))) int      v8i;
typedef __attribute__((ext_vector_type(4))) int      v4i;
typedef __attribute__((address_space(1))) v4i*       as1_v4i;   // global <4 x i32>*
typedef __attribute__((address_space(3))) v4i*       as3_v4i;   // LDS    <4 x i32>*

#define TOK   131072      // B*H*W = 8*128*128
#define DIM   256
#define IMH   128
#define IMW   128
#define NB    8
#define HEADS 4
#define HDIM  64
#define WS    4
#define DISP  2
#define NWIN  1024        // (128/4)*(128/4)

static __device__ inline __bf16 f2bf(float f) {
  union { float f; unsigned u; } a; a.f = f;
  unsigned r = a.u + 0x7FFFu + ((a.u >> 16) & 1u);   // round-to-nearest-even
  union { unsigned short s; __bf16 b; } o; o.s = (unsigned short)(r >> 16);
  return o.b;
}
static __device__ inline float bf2f(__bf16 b) {
  union { __bf16 b; unsigned short s; } i; i.b = b;
  union { unsigned u; float f; } o; o.u = ((unsigned)i.s) << 16;
  return o.f;
}

// ---------------- CDNA5 async / TDM helpers (guarded) ----------------
#if __has_builtin(__builtin_amdgcn_global_load_async_to_lds_b128)
#define HAS_ASYNC 1
static __device__ inline void async_cp_b128(void* lds, const void* gm) {
  __builtin_amdgcn_global_load_async_to_lds_b128(
      (as1_v4i)(uintptr_t)gm, (as3_v4i)(unsigned)(uintptr_t)lds, 0, 0);
}
#endif

static __device__ inline void wait_async0() {
#if __has_builtin(__builtin_amdgcn_s_wait_asynccnt)
  __builtin_amdgcn_s_wait_asynccnt(0);
#else
  asm volatile("s_wait_asynccnt 0" ::: "memory");
#endif
}
static __device__ inline void wait_tensor0() {
#if __has_builtin(__builtin_amdgcn_s_wait_tensorcnt)
  __builtin_amdgcn_s_wait_tensorcnt(0);
#else
  asm volatile("s_wait_tensorcnt 0" ::: "memory");
#endif
}

#if __has_builtin(__builtin_amdgcn_tensor_load_to_lds)
#define HAS_TDM 1
// 2D bf16 tile DMA: rows x cols tile from row-major tensor (stride elems/row).
static __device__ inline void tdm_load_tile_bf16(const __bf16* gsrc, unsigned lds_off,
                                                 int rows, int cols, int stride_elems,
                                                 int tdim0, int tdim1) {
  unsigned long long ga = (unsigned long long)(uintptr_t)gsrc;
  v4u g0;
  g0.x = 1u;                                            // count=1 (user D#)
  g0.y = lds_off;                                       // lds_addr (bytes)
  g0.z = (unsigned)(ga & 0xFFFFFFFFu);                  // global_addr[31:0]
  g0.w = (unsigned)((ga >> 32) & 0x01FFFFFFu) | (2u << 30);  // addr[56:32] | type=2
  v8i g1;
  g1[0] = (int)(1u << 16);                              // data_size=1 (2 bytes)
  g1[1] = (int)(((unsigned)tdim0 & 0xFFFFu) << 16);     // tensor_dim0[15:0]
  g1[2] = (int)((((unsigned)tdim0 >> 16) & 0xFFFFu) |
                (((unsigned)tdim1 & 0xFFFFu) << 16));   // tensor_dim0[31:16] | dim1[15:0]
  g1[3] = (int)((((unsigned)tdim1 >> 16) & 0xFFFFu) |
                (((unsigned)cols & 0xFFFFu) << 16));    // tensor_dim1[31:16] | tile_dim0
  g1[4] = rows;                                         // tile_dim1 (tile_dim2 = 0)
  g1[5] = stride_elems;                                 // tensor_dim0_stride[31:0]
  g1[6] = 0;
  g1[7] = 0;
  v4i z4 = {0, 0, 0, 0};
#if __clang_major__ >= 23
  v8i z8 = {0, 0, 0, 0, 0, 0, 0, 0};
  __builtin_amdgcn_tensor_load_to_lds(g0, g1, z4, z4, z8, 0);
#else
  __builtin_amdgcn_tensor_load_to_lds(g0, g1, z4, z4, 0);
#endif
}
#endif

// ---------------------------------------------------------------------------
// Tiled GEMM:  C[M,N](bf16) = epi(A[M,K](bf16) @ W[K,N](bf16) + bias(f32))
// 256 threads = 8 waves; tile 64x32; wave -> one 16x16 WMMA tile; K step 32.
// A tile staged by TDM (wave 0); B tile staged transposed for contiguous frags.
// EPI: 0 = bias (may be null), 1 = bias+GELU(exact), 2 = bias+residual(bf16).
// ---------------------------------------------------------------------------
template<int EPI>
__global__ __launch_bounds__(256) void gemm_bf16_wmma(
    const __bf16* __restrict__ A, const __bf16* __restrict__ W,
    const float* __restrict__ bias, const __bf16* __restrict__ R,
    __bf16* __restrict__ C, int M, int N, int K)
{
  __shared__ __bf16 As[64][32];      // row-major MxK tile (TDM destination)
  __shared__ __bf16 BsT[32][32];     // transposed: [n][k]
  const int tid  = threadIdx.x;
  const int lane = tid & 31;
  const int wave = tid >> 5;
  const int wm = wave & 3, wn = wave >> 2;
  const int m0 = blockIdx.y * 64;
  const int n0 = blockIdx.x * 32;

  const int mloc  = wm * 16 + (lane & 15);
  const int nloc  = wn * 16 + (lane & 15);
  const int colC  = n0 + nloc;
  const int koff  = (lane >> 4) * 8;   // A-frag K offset (ISA 16x32 layout)
  const int kbase = (lane >> 4) * 16;  // unused: B frag also indexes with koff
#if defined(HAS_TDM)
  const unsigned ldsAoff = (unsigned)(uintptr_t)&As[0][0];
#endif

  v8f acc = {};
  for (int kk = 0; kk < K; kk += 32) {
    // ---- A tile: 64x32 bf16 via Tensor Data Mover (single DMA, wave 0) ----
#if defined(HAS_TDM)
    if (wave == 0)
      tdm_load_tile_bf16(A + (size_t)m0 * K + kk, ldsAoff, 64, 32, K, K, M);
#else
    for (int i = tid; i < 64 * 32 / 8; i += 256) {   // 16B per thread fallback
      int r = i >> 2, c = (i & 3) * 8;
      *(ulonglong2*)&As[r][c] =
          *(const ulonglong2*)(A + (size_t)(m0 + r) * K + kk + c);
    }
#endif
    // ---- B tile: 32x32 bf16, stored transposed (coalesced global reads) ----
    for (int i = tid; i < 32 * 32; i += 256) {
      int r = i >> 5, c = i & 31;                    // r = k, c = n
      BsT[c][r] = W[(size_t)(kk + r) * N + n0 + c];
    }
#if defined(HAS_TDM)
    if (wave == 0) wait_tensor0();
#endif
    __syncthreads();

    v16bf af, bf;
#pragma unroll
    for (int i = 0; i < 8; ++i) {
      af[i]     = As[mloc][koff + i];
      af[i + 8] = As[mloc][16 + koff + i];
    }
#pragma unroll
    for (int i = 0; i < 8; ++i) {
      bf[i]     = BsT[nloc & 31][koff + i];
      bf[i + 8] = BsT[nloc & 31][16 + koff + i];
    }
    acc = __builtin_amdgcn_wmma_f32_16x16x32_bf16(false, af, false, bf,
                                                  (short)0, acc, false, false);
    __syncthreads();
  }

#pragma unroll
  for (int r = 0; r < 8; ++r) {
    int row = m0 + wm * 16 + r + 8 * (lane >> 4);
    float v = acc[r];
    if (bias) v += bias[colC];
    if (EPI == 1) v = 0.5f * v * (1.0f + erff(v * 0.70710678118654752f));
    if (EPI == 2) v += bf2f(R[(size_t)row * N + colC]);
    C[(size_t)row * N + colC] = f2bf(v);
  }
}

// ---------------------------------------------------------------------------
// Window attention: one wave per (batch, head, window). q/k/v gathered into
// LDS with async-to-LDS copies. scores = q@k^T (2 WMMAs), +pos/+masks,
// softmax, out = attn@v (4 WMMAs, K zero-padded 16->32). Roll folded in.
// ---------------------------------------------------------------------------
__global__ __launch_bounds__(128) void win_attn_wmma(
    const __bf16* __restrict__ Q, int ldq,
    const __bf16* __restrict__ Kp, int ldk,
    const __bf16* __restrict__ V, int ldv,
    const float* __restrict__ pos,     // [7,7]
    __bf16* __restrict__ O,            // [TOK, 256]
    int shifted)
{
  __shared__ __bf16 qs[4][16][64];
  __shared__ __bf16 ks[4][16][64];
  __shared__ __bf16 vs[4][32][64];     // rows 16..31 zero (K padding)
  __shared__ __bf16 am[4][16][32];     // cols 16..31 zero (K padding)
  __shared__ float  sc[4][16][16];

  const int lane = threadIdx.x & 31;
  const int wave = threadIdx.x >> 5;
  const int gid  = blockIdx.x * 4 + wave;        // 0 .. 32767
  const int win  = gid & (NWIN - 1);
  const int bh   = gid >> 10;
  const int head = bh & 3;
  const int b    = bh >> 2;
  const int wy   = win >> 5, wx = win & 31;

  auto tok = [&](int t) -> int {                 // window cell -> token index
    int ti = t >> 2, tj = t & 3;
    int y = wy * 4 + ti, x = wx * 4 + tj;
    if (shifted) { y = (y + DISP) & (IMH - 1); x = (x + DISP) & (IMW - 1); }
    return (b * IMH + y) * IMW + x;
  };

  // gather q,k,v (roll folded into token index)
#if defined(HAS_ASYNC)
  for (int i = lane; i < 16 * 8; i += 32) {      // 16B chunks: 8 per token row
    int t = i >> 3, c = (i & 7) * 8;
    size_t tk = (size_t)tok(t);
    async_cp_b128(&qs[wave][t][c], Q  + tk * ldq + head * HDIM + c);
    async_cp_b128(&ks[wave][t][c], Kp + tk * ldk + head * HDIM + c);
    async_cp_b128(&vs[wave][t][c], V  + tk * ldv + head * HDIM + c);
  }
#else
  for (int i = lane; i < 16 * 64; i += 32) {
    int t = i >> 6, d = i & 63;
    size_t tk = (size_t)tok(t);
    qs[wave][t][d] = Q [tk * ldq + head * HDIM + d];
    ks[wave][t][d] = Kp[tk * ldk + head * HDIM + d];
    vs[wave][t][d] = V [tk * ldv + head * HDIM + d];
  }
#endif
  for (int i = lane; i < 16 * 64; i += 32)       // zero K padding of v
    vs[wave][16 + (i >> 6)][i & 63] = f2bf(0.0f);
  for (int i = lane; i < 16 * 16; i += 32)       // zero K padding of attn
    am[wave][i >> 4][16 + (i & 15)] = f2bf(0.0f);
#if defined(HAS_ASYNC)
  wait_async0();
#endif
  __syncthreads();

  const int koff  = (lane >> 4) * 8;
  const int kbase = (lane >> 4) * 16;
  const int nloc  = lane & 15;

  // scores = q @ k^T   (A = q 16x64, B[k][n] = ks[n][k])
  v8f s = {};
#pragma unroll
  for (int kc = 0; kc < 64; kc += 32) {
    v16bf a, bm;
#pragma unroll
    for (int i = 0; i < 8; ++i) {
      a[i]     = qs[wave][nloc][kc + koff + i];
      a[i + 8] = qs[wave][nloc][kc + 16 + koff + i];
    }
#pragma unroll
    for (int i = 0; i < 8; ++i) {
      bm[i]     = ks[wave][nloc][kc + koff + i];
      bm[i + 8] = ks[wave][nloc][kc + 16 + koff + i];
    }
    s = __builtin_amdgcn_wmma_f32_16x16x32_bf16(false, a, false, bm,
                                                (short)0, s, false, false);
  }

  // scale + relative-position bias + shifted-window masks -> LDS scores
#pragma unroll
  for (int r = 0; r < 8; ++r) {
    int irow = r + 8 * (lane >> 4);
    int j = nloc;
    int ti = irow >> 2, tj = irow & 3, tk = j >> 2, tl = j & 3;
    float v = s[r] * 0.125f + pos[(ti - tk + 3) * 7 + (tj - tl + 3)];
    if (shifted) {
      if (wy == 31 && ((ti >= 2) != (tk >= 2))) v = -1e30f;  // UL mask, bottom row
      if (wx == 31 && ((tj >= 2) != (tl >= 2))) v = -1e30f;  // LR mask, right col
    }
    sc[wave][irow][j] = v;
  }
  __syncthreads();

  // per-row softmax (lanes 0..15, one row each)
  if (lane < 16) {
    float mx = -1e38f;
#pragma unroll
    for (int j = 0; j < 16; ++j) mx = fmaxf(mx, sc[wave][lane][j]);
    float sum = 0.0f;
    float e[16];
#pragma unroll
    for (int j = 0; j < 16; ++j) { e[j] = __expf(sc[wave][lane][j] - mx); sum += e[j]; }
    float inv = 1.0f / sum;
#pragma unroll
    for (int j = 0; j < 16; ++j) am[wave][lane][j] = f2bf(e[j] * inv);
  }
  __syncthreads();

  // out = attn @ v ; 4 N-tiles of 16
#pragma unroll
  for (int nt = 0; nt < 4; ++nt) {
    v16bf a, bm;
#pragma unroll
    for (int i = 0; i < 8; ++i) {
      a[i]     = am[wave][nloc][koff + i];
      a[i + 8] = am[wave][nloc][16 + koff + i];
    }
#pragma unroll
    for (int i = 0; i < 8; ++i) {
      bm[i]     = vs[wave][koff + i][nt * 16 + nloc];
      bm[i + 8] = vs[wave][16 + koff + i][nt * 16 + nloc];
    }
    v8f o = {};
    o = __builtin_amdgcn_wmma_f32_16x16x32_bf16(false, a, false, bm,
                                                (short)0, o, false, false);
#pragma unroll
    for (int r = 0; r < 8; ++r) {
      int irow = r + 8 * (lane >> 4);
      O[(size_t)tok(irow) * 256 + head * HDIM + nt * 16 + nloc] = f2bf(o[r]);
    }
  }
}

// ---------------------------------------------------------------------------
// LayerNorm over 256 features (bf16 in/out, f32 math): one wave per token.
// ---------------------------------------------------------------------------
__global__ __launch_bounds__(256) void layernorm_k(
    const __bf16* __restrict__ X, const float* __restrict__ g,
    const float* __restrict__ bta, __bf16* __restrict__ Y, int ntok)
{
  int wave = threadIdx.x >> 5, lane = threadIdx.x & 31;
  int t = blockIdx.x * 8 + wave;
  if (t >= ntok) return;
  const __bf16* x = X + (size_t)t * DIM;
  float v[8], s = 0.0f;
#pragma unroll
  for (int i = 0; i < 8; ++i) { v[i] = bf2f(x[lane + i * 32]); s += v[i]; }
#pragma unroll
  for (int off = 16; off; off >>= 1) s += __shfl_xor(s, off, 32);
  float mean = s * (1.0f / DIM);
  float var = 0.0f;
#pragma unroll
  for (int i = 0; i < 8; ++i) { float d = v[i] - mean; var += d * d; }
#pragma unroll
  for (int off = 16; off; off >>= 1) var += __shfl_xor(var, off, 32);
  float rs = rsqrtf(var * (1.0f / DIM) + 1e-5f);
#pragma unroll
  for (int i = 0; i < 8; ++i) {
    int c = lane + i * 32;
    Y[(size_t)t * DIM + c] = f2bf((v[i] - mean) * rs * g[c] + bta[c]);
  }
}

// ---------------------------------------------------------------------------
// NCHW f32 -> token-major bf16, and back (32x32 LDS tiles, 256 threads)
// ---------------------------------------------------------------------------
__global__ __launch_bounds__(256) void nchw2tok(const float* __restrict__ in,
                                                __bf16* __restrict__ out)
{
  __shared__ float tile[32][33];
  int b = blockIdx.z;
  int p0 = blockIdx.x * 32, c0 = blockIdx.y * 32;
  int tx = threadIdx.x & 31, ty = threadIdx.x >> 5;
  for (int r = ty; r < 32; r += 8)
    tile[r][tx] = in[((size_t)b * DIM + c0 + r) * (IMH * IMW) + p0 + tx];
  __syncthreads();
  for (int r = ty; r < 32; r += 8)
    out[((size_t)b * (IMH * IMW) + p0 + r) * DIM + c0 + tx] = f2bf(tile[tx][r]);
}

__global__ __launch_bounds__(256) void tok2nchw(const __bf16* __restrict__ in,
                                                float* __restrict__ out)
{
  __shared__ float tile[32][33];
  int b = blockIdx.z;
  int p0 = blockIdx.x * 32, c0 = blockIdx.y * 32;
  int tx = threadIdx.x & 31, ty = threadIdx.x >> 5;
  for (int r = ty; r < 32; r += 8)
    tile[r][tx] = bf2f(in[((size_t)b * (IMH * IMW) + p0 + r) * DIM + c0 + tx]);
  __syncthreads();
  for (int r = ty; r < 32; r += 8)
    out[((size_t)b * DIM + c0 + r) * (IMH * IMW) + p0 + tx] = tile[tx][r];
}

__global__ __launch_bounds__(256) void cvt_f32_bf16(const float* __restrict__ in,
                                                    __bf16* __restrict__ out, int n)
{
  int i = blockIdx.x * 256 + threadIdx.x;
  if (i < n) out[i] = f2bf(in[i]);
}

// ---------------------------------------------------------------------------
// Host orchestration
// ---------------------------------------------------------------------------
struct Blk {
  const float *ln1_g, *ln1_b, *pos, *bo, *ln2_g, *ln2_b, *b1, *b2;
  const __bf16 *wo, *w1, *w2, *wq, *wkv, *wqkv;
};

extern "C" void kernel_launch(void* const* d_in, const int* in_sizes, int n_in,
                              void* d_out, int out_size, void* d_ws, size_t ws_size,
                              hipStream_t stream)
{
  (void)in_sizes; (void)n_in; (void)out_size; (void)ws_size;
  const float* feature      = (const float*)d_in[0];
  const float* feature_edge = (const float*)d_in[1];
  int pi = 2;
  auto nxt = [&]() { return (const float*)d_in[pi++]; };

  // workspace layout (bf16)
  __bf16* bws = (__bf16*)d_ws;
  const size_t TD = (size_t)TOK * DIM;
  __bf16* t0  = bws;                        // ping
  __bf16* t1  = t0  + TD;                   // pong
  __bf16* y0  = t1  + TD;                   // cross query source
  __bf16* ln  = y0  + TD;                   // LN output / staging
  __bf16* qkv = ln  + TD;                   // TOK x 768
  __bf16* att = qkv + (size_t)TOK * 768;    // attention out
  __bf16* x1  = att + TD;                   // post-attention residual
  __bf16* hb  = x1  + TD;                   // TOK x 1024 MLP hidden
  __bf16* warena = hb + (size_t)TOK * 1024; // bf16 weight arena

  auto cvtW = [&](const float* src, size_t n) -> const __bf16* {
    __bf16* dst = warena; warena += n;
    cvt_f32_bf16<<<(unsigned)((n + 255) / 256), 256, 0, stream>>>(src, dst, (int)n);
    return dst;
  };
  auto loadBlk = [&](bool cross) {
    Blk k{};
    k.ln1_g = nxt(); k.ln1_b = nxt(); k.pos = nxt();
    const float* wo = nxt(); k.bo = nxt();
    k.ln2_g = nxt(); k.ln2_b = nxt();
    const float* w1 = nxt(); k.b1 = nxt();
    const float* w2 = nxt(); k.b2 = nxt();
    k.wo = cvtW(wo, 256 * 256);
    k.w1 = cvtW(w1, 256 * 1024);
    k.w2 = cvtW(w2, 1024 * 256);
    if (cross) { k.wq = cvtW(nxt(), 256 * 256); k.wkv = cvtW(nxt(), 256 * 512); }
    else       { k.wqkv = cvtW(nxt(), 256 * 768); }
    return k;
  };

  const float* c_pm_w_f = nxt(); const float* c_pm_b = nxt();
  Blk cb0 = loadBlk(true), cb1 = loadBlk(true);
  const float* s_pm_w_f = nxt(); const float* s_pm_b = nxt();
  Blk sb0 = loadBlk(false), sb1 = loadBlk(false);
  const __bf16* c_pm_w = cvtW(c_pm_w_f, 256 * 256);
  const __bf16* s_pm_w = cvtW(s_pm_w_f, 256 * 256);

  auto gemm = [&](const __bf16* A, const __bf16* Wt, const float* bias,
                  const __bf16* R, __bf16* C, int M, int N, int K, int epi) {
    dim3 g(N / 32, M / 64);
    if (epi == 0)      gemm_bf16_wmma<0><<<g, 256, 0, stream>>>(A, Wt, bias, R, C, M, N, K);
    else if (epi == 1) gemm_bf16_wmma<1><<<g, 256, 0, stream>>>(A, Wt, bias, R, C, M, N, K);
    else               gemm_bf16_wmma<2><<<g, 256, 0, stream>>>(A, Wt, bias, R, C, M, N, K);
  };

  auto runBlock = [&](__bf16* xin, __bf16* xout, const Blk& bk,
                      bool shifted, bool cross, const __bf16* ybuf) {
    layernorm_k<<<TOK / 8, 256, 0, stream>>>(xin, bk.ln1_g, bk.ln1_b, ln, TOK);
    const __bf16 *qp, *kp, *vp; int ldq, ldk, ldv;
    if (cross) {
      gemm(ybuf, bk.wq,  nullptr, nullptr, qkv,      TOK, 256, 256, 0);
      gemm(ln,   bk.wkv, nullptr, nullptr, qkv + TD, TOK, 512, 256, 0);
      qp = qkv;            ldq = 256;
      kp = qkv + TD;       ldk = 512;
      vp = qkv + TD + 256; ldv = 512;
    } else {
      gemm(ln, bk.wqkv, nullptr, nullptr, qkv, TOK, 768, 256, 0);
      qp = qkv;       ldq = 768;
      kp = qkv + 256; ldk = 768;
      vp = qkv + 512; ldv = 768;
    }
    win_attn_wmma<<<(NB * HEADS * NWIN) / 4, 128, 0, stream>>>(
        qp, ldq, kp, ldk, vp, ldv, bk.pos, att, shifted ? 1 : 0);
    gemm(att, bk.wo, bk.bo, xin, x1, TOK, 256, 256, 2);         // + residual
    layernorm_k<<<TOK / 8, 256, 0, stream>>>(x1, bk.ln2_g, bk.ln2_b, ln, TOK);
    gemm(ln, bk.w1, bk.b1, nullptr, hb, TOK, 1024, 256, 1);     // GELU
    gemm(hb, bk.w2, bk.b2, x1, xout, TOK, 256, 1024, 2);        // + residual
  };

  dim3 tg(512, 8, NB);
  // ---- cross module ----
  nchw2tok<<<tg, 256, 0, stream>>>(feature, t0);
  nchw2tok<<<tg, 256, 0, stream>>>(feature_edge, ln);  // stage y tokens
  gemm(ln, c_pm_w, c_pm_b, nullptr, y0, TOK, 256, 256, 0);
  gemm(t0, c_pm_w, c_pm_b, nullptr, t1, TOK, 256, 256, 0);
  runBlock(t1, t0, cb0, false, true, y0);
  runBlock(t0, t1, cb1, true,  true, y0);
  // ---- self module ----
  gemm(t1, s_pm_w, s_pm_b, nullptr, t0, TOK, 256, 256, 0);
  runBlock(t0, t1, sb0, false, false, nullptr);
  runBlock(t1, t0, sb1, true,  false, nullptr);
  // ---- output NCHW ----
  tok2nchw<<<tg, 256, 0, stream>>>(t0, (float*)d_out);
}
